// DecisionTransformer_30915174596632
// MI455X (gfx1250) — matrix operand order repaired and checked by reading, not compiled, not run on hardware
//
#include <hip/hip_runtime.h>

// ---------------------------------------------------------------------------
// Decision Transformer forward for MI455X (gfx1250, wave32, WMMA + TDM).
// GEMMs: v_wmma_f32_16x16x32_bf16 with double-buffered tensor_load_to_lds
// (Tensor Data Mover) tile staging, gated by s_wait_tensorcnt.
// Attention: flash-style streaming softmax, block-cooperative K/V staging
// (K via TDM, V via transposed LDS stage), WMMA for QK^T and PV.
// Workspace requirement: ~512 MB.
// ---------------------------------------------------------------------------

typedef __attribute__((ext_vector_type(16))) __bf16       v16bf;
typedef __attribute__((ext_vector_type(8)))  float        v8f;
typedef __attribute__((ext_vector_type(4)))  unsigned int u32x4;
typedef __attribute__((ext_vector_type(8)))  int          i32x8;
typedef __attribute__((ext_vector_type(4)))  int          i32x4;

union BF16x16 { v16bf v; u32x4 q[2]; };

#define DEV static __device__ __forceinline__

constexpr int Bsz  = 16;
constexpr int SEQ  = 256;
constexpr int STATE= 128;
constexpr int ACTN = 18;
constexpr int Dm   = 1024;
constexpr int FFN  = 4096;
constexpr int Hn   = 16;
constexpr int NBLK = 6;
constexpr int Tt   = 3 * SEQ;     // 768 tokens
constexpr int HD   = Dm / Hn;     // 64
constexpr int Mtok = Bsz * Tt;    // 12288 rows
constexpr int BH   = Bsz * Hn;    // 256 (b,h) pairs

DEV v8f wmma_bf16(v16bf a, v16bf b, v8f c) {
    return __builtin_amdgcn_wmma_f32_16x16x32_bf16(
        /*neg_a=*/false, a, /*neg_b=*/false, b,
        /*c_mod=*/(short)0, c, /*reuse_a=*/false, /*reuse_b=*/false);
}

DEV void put8(v16bf& d, int base, float4 a, float4 b) {
    d[base + 0] = (__bf16)a.x; d[base + 1] = (__bf16)a.y;
    d[base + 2] = (__bf16)a.z; d[base + 3] = (__bf16)a.w;
    d[base + 4] = (__bf16)b.x; d[base + 5] = (__bf16)b.y;
    d[base + 6] = (__bf16)b.z; d[base + 7] = (__bf16)b.w;
}

// ---------------------------------------------------------------------------
// Tensor Data Mover: 2D tile load, global -> LDS.
// D# packed per CDNA5 ISA (groups 2/3 zero => 2D).
// data_size_code: 0=1B,1=2B,2=4B,3=8B. Dims/strides in elements.
// 6-arg builtin form (this toolchain): (g0, g1, g2, g3, g4, cpol).
// ---------------------------------------------------------------------------
DEV void tdm_load_2d(unsigned lds_off, const void* gptr,
                     unsigned data_size_code,
                     unsigned tile_x, unsigned tile_y,
                     unsigned tensor_d0, unsigned tensor_d1,
                     unsigned long long stride0)
{
    unsigned long long ga = (unsigned long long)gptr;
    u32x4 g0;
    g0[0] = 1u;                                   // count=1, user descriptor
    g0[1] = lds_off;                              // LDS byte address
    g0[2] = (unsigned)ga;                         // global_addr[31:0]
    g0[3] = (unsigned)((ga >> 32) & 0x01FFFFFFull) | 0x80000000u; // type=2
    i32x8 g1;
    g1[0] = (int)(data_size_code << 16);          // mask=0, no pad/iterate
    g1[1] = (int)((tensor_d0 & 0xFFFFu) << 16);   // barrier_addr=0 | d0.lo
    g1[2] = (int)(((tensor_d0 >> 16) & 0xFFFFu) | ((tensor_d1 & 0xFFFFu) << 16));
    g1[3] = (int)(((tensor_d1 >> 16) & 0xFFFFu) | ((tile_x & 0xFFFFu) << 16));
    g1[4] = (int)(tile_y & 0xFFFFu);              // tile_dim1 | tile_dim2=0
    g1[5] = (int)(stride0 & 0xFFFFFFFFull);       // dim0 stride lo
    g1[6] = (int)((stride0 >> 32) & 0xFFFFull);   // dim0 stride hi | s1.lo=0
    g1[7] = 0;
    i32x4 z4 = {0, 0, 0, 0};
    i32x8 z8 = {0, 0, 0, 0, 0, 0, 0, 0};
    __builtin_amdgcn_tensor_load_to_lds(g0, g1, z4, z4, z8, 0);
}

DEV unsigned lds_addr_of(const void* p) {
    return (unsigned)(unsigned long long)p;       // LDS aperture: addr[31:0]
}

// ---------------------------------------------------------------------------
// Embedding: x[b, 3s+{0,1,2}, d] = tanh(tok_embed) + global_pos + pos_emb
// ---------------------------------------------------------------------------
__global__ __launch_bounds__(256) void embed_kernel(
    const float* __restrict__ state, const int* __restrict__ action,
    const float* __restrict__ rtg, const int* __restrict__ tstep,
    const float* __restrict__ wS, const float* __restrict__ bS,
    const float* __restrict__ embA,
    const float* __restrict__ wR, const float* __restrict__ bR,
    const float* __restrict__ pos_emb, const float* __restrict__ gpe,
    float* __restrict__ X)
{
    int idx = blockIdx.x * blockDim.x + threadIdx.x;  // over B*SEQ*D
    if (idx >= Bsz * SEQ * Dm) return;
    int d = idx % Dm;
    int s = (idx / Dm) % SEQ;
    int b = idx / (Dm * SEQ);

    const float inv255 = 1.0f / 255.0f;
    float acc = bS[d];
    const float* sp = state + (size_t)(b * SEQ + s) * STATE;
#pragma unroll 4
    for (int k = 0; k < STATE; ++k) acc += sp[k] * inv255 * wS[k * Dm + d];
    float es = tanhf(acc);
    float er = tanhf(rtg[b * SEQ + s] * wR[d] + bR[d]);
    int a    = action[b * SEQ + s];
    float ea = tanhf(embA[(size_t)a * Dm + d]);

    int ts  = tstep[b];
    float g = gpe[(size_t)ts * Dm + d];

    float* xb = X + (size_t)(b * Tt + 3 * s) * Dm + d;
    xb[0]      = er + g + pos_emb[(size_t)(3 * s + 0) * Dm + d];
    xb[Dm]     = es + g + pos_emb[(size_t)(3 * s + 1) * Dm + d];
    xb[2 * Dm] = ea + g + pos_emb[(size_t)(3 * s + 2) * Dm + d];
}

// ---------------------------------------------------------------------------
// fp32 [K][N] -> bf16 transposed [N][K] weight staging (so TDM lands the
// GEMM B tile directly in the [n][k] LDS layout -> vectorized fragments).
// ---------------------------------------------------------------------------
__global__ __launch_bounds__(256) void cvt_bf16_t_kernel(
    const float* __restrict__ src, __bf16* __restrict__ dst, int K, int N)
{
    int idx = blockIdx.x * blockDim.x + threadIdx.x;
    int stride = gridDim.x * blockDim.x;
    int total = K * N;
    for (; idx < total; idx += stride) {
        int k = idx / N, n = idx % N;                 // coalesced read
        dst[(size_t)n * K + k] = (__bf16)src[idx];    // transposed write
    }
}

// ---------------------------------------------------------------------------
// GEMM: C[M,N] = act(A[M,K] @ W + bias); A fp32, W given transposed bf16
// Wt[N][K]. 128x128 tile / workgroup, wave tile 64x32, K-step 32.
// Double-buffered TDM tile staging issued by wave 0.
// ---------------------------------------------------------------------------
template <int RELU>
__global__ __launch_bounds__(256) void gemm_wmma_kernel(
    const float* __restrict__ A, const __bf16* __restrict__ Wt,
    const float* __restrict__ bias, float* __restrict__ C,
    int M, int N, int K)
{
    __shared__ float  Af[2][128][32];   // A tile fp32 [m][k], 16KB each
    __shared__ __bf16 Bt[2][128][32];   // B tile bf16 [n][k],  8KB each

    const int tid  = threadIdx.x;
    const int lane = tid & 31;
    const int w    = tid >> 5;       // 0..7
    const int wm   = w >> 2;         // 0..1
    const int wn   = w & 3;          // 0..3
    const int m0   = blockIdx.y * 128;
    const int n0   = blockIdx.x * 128;

    const int laneM = lane & 15;
    const int kbA   = (lane < 16) ? 0 : 8;   // 16-bit A-fragment K base
    const int kbB   = (lane < 16) ? 0 : 8;   // B-fragment K base

    const float*  Abase = A  + (size_t)m0 * K;
    const __bf16* Bbase = Wt + (size_t)n0 * K;
    const int ntiles = K >> 5;

    if (w == 0) {
        tdm_load_2d(lds_addr_of(&Af[0][0][0]), Abase, 2u, 32u, 128u, 32u, 128u,
                    (unsigned long long)K);
        tdm_load_2d(lds_addr_of(&Bt[0][0][0]), Bbase, 1u, 32u, 128u, 32u, 128u,
                    (unsigned long long)K);
    }

    v8f acc[4][2] = {};

    for (int it = 0; it < ntiles; ++it) {
        const int buf = it & 1;
        __syncthreads();                       // buf^1 free (tile it-1 consumed)
        if (w == 0) {
            if (it + 1 < ntiles) {
                const int kn = (it + 1) * 32;
                tdm_load_2d(lds_addr_of(&Af[buf ^ 1][0][0]), Abase + kn,
                            2u, 32u, 128u, 32u, 128u, (unsigned long long)K);
                tdm_load_2d(lds_addr_of(&Bt[buf ^ 1][0][0]), Bbase + kn,
                            1u, 32u, 128u, 32u, 128u, (unsigned long long)K);
                __builtin_amdgcn_s_wait_tensorcnt(2);  // tile 'it' complete
            } else {
                __builtin_amdgcn_s_wait_tensorcnt(0);
            }
        }
        __syncthreads();                       // tile 'it' visible

        v16bf af[4], bfrg[2];
#pragma unroll
        for (int mi = 0; mi < 4; ++mi) {
            const float* p = &Af[buf][wm * 64 + mi * 16 + laneM][kbA];
            float4 x0 = *(const float4*)(p);
            float4 x1 = *(const float4*)(p + 4);
            float4 x2 = *(const float4*)(p + 16);
            float4 x3 = *(const float4*)(p + 20);
            BF16x16 t;
            put8(t.v, 0, x0, x1);
            put8(t.v, 8, x2, x3);
            af[mi] = t.v;
        }
#pragma unroll
        for (int ni = 0; ni < 2; ++ni) {
            const __bf16* p = &Bt[buf][wn * 32 + ni * 16 + laneM][kbB];
            BF16x16 t;
            t.q[0] = *(const u32x4*)p;
            t.q[1] = *(const u32x4*)(p + 16);
            bfrg[ni] = t.v;
        }
#pragma unroll
        for (int mi = 0; mi < 4; ++mi)
#pragma unroll
            for (int ni = 0; ni < 2; ++ni)
                acc[mi][ni] = wmma_bf16(af[mi], bfrg[ni], acc[mi][ni]);
    }

    // ---- epilogue: bias (+relu), fp32 store ----
    const int mrowbase = m0 + wm * 64;
    const int colbase  = n0 + wn * 32;
    const int mhalf    = (lane < 16) ? 0 : 8;
#pragma unroll
    for (int ni = 0; ni < 2; ++ni) {
        int col  = colbase + ni * 16 + laneM;
        float bv = bias ? bias[col] : 0.0f;
#pragma unroll
        for (int mi = 0; mi < 4; ++mi) {
#pragma unroll
            for (int r = 0; r < 8; ++r) {
                int row  = mrowbase + mi * 16 + r + mhalf;
                float vv = acc[mi][ni][r] + bv;
                if (RELU) vv = fmaxf(vv, 0.0f);
                C[(size_t)row * N + col] = vv;
            }
        }
    }
}

// ---------------------------------------------------------------------------
// Causal attention, no 1/sqrt(d) scaling (faithful to reference).
// Block owns 128 q-rows (8 waves x 16); all waves run the same k-step count
// (fully-masked trailing steps are exact numeric no-ops: alpha=1, p=0), so
// K/V tiles are staged cooperatively: K tile via TDM (contiguous 8KB),
// V tile transposed to bf16 in LDS for vectorized B-fragments.
// Q/K/V/O are [BH, T, HD] flat (reference's raw head reshape = free reinterp).
// ---------------------------------------------------------------------------
__global__ __launch_bounds__(256) void attn_kernel(
    const float* __restrict__ Qm, const float* __restrict__ Km,
    const float* __restrict__ Vm, float* __restrict__ Om)
{
    __shared__ float  Ksf[32][64];      // K tile fp32 via TDM (8KB)
    __shared__ __bf16 Vt[64][32];       // V tile bf16 transposed [hd][k] (4KB)
    __shared__ __bf16 Sp[8][16][32];    // per-wave P staging (8KB)

    const int tid   = threadIdx.x;
    const int lane  = tid & 31;
    const int w     = tid >> 5;
    const int bh    = blockIdx.x;                   // 0..255
    const int q0    = blockIdx.y * 128 + w * 16;    // wave's q-tile
    const int laneM = lane & 15;
    const int hi    = (lane < 16) ? 0 : 1;

    const float* qp = Qm + (size_t)bh * Tt * HD;
    const float* kp = Km + (size_t)bh * Tt * HD;
    const float* vp = Vm + (size_t)bh * Tt * HD;

    // Q fragments (HD=64 -> two K=32 A-fragments), fp32 -> bf16
    v16bf qf[2];
#pragma unroll
    for (int f = 0; f < 2; ++f) {
        const float* p = qp + (size_t)(q0 + laneM) * HD + f * 32 + hi * 8;
        float4 x0 = *(const float4*)(p);
        float4 x1 = *(const float4*)(p + 4);
        float4 x2 = *(const float4*)(p + 16);
        float4 x3 = *(const float4*)(p + 20);
        BF16x16 t;
        put8(t.v, 0, x0, x1);
        put8(t.v, 8, x2, x3);
        qf[f] = t.v;
    }

    float mrow[8], lrow[8];
#pragma unroll
    for (int r = 0; r < 8; ++r) { mrow[r] = -3.0e38f; lrow[r] = 0.0f; }
    v8f o[4] = {};

    // V staging assignment: thread -> (row, 8 cols)
    const int vr  = tid >> 3;          // 0..31
    const int vc0 = (tid & 7) * 8;     // 0..56

    const int ksteps = blockIdx.y * 4 + 4;   // covers k <= blockIdx.y*128+127
    for (int it = 0; it < ksteps; ++it) {
        const int j0 = it * 32;
        __syncthreads();                       // previous tile consumed
        if (w == 0)
            tdm_load_2d(lds_addr_of(&Ksf[0][0]), kp + (size_t)j0 * HD,
                        2u, 2048u, 1u, 2048u, 1u, 2048ull);
        // V tile -> transposed bf16 LDS
        {
            const float* vrow = vp + (size_t)(j0 + vr) * HD + vc0;
            float4 a = *(const float4*)(vrow);
            float4 b = *(const float4*)(vrow + 4);
            Vt[vc0 + 0][vr] = (__bf16)a.x; Vt[vc0 + 1][vr] = (__bf16)a.y;
            Vt[vc0 + 2][vr] = (__bf16)a.z; Vt[vc0 + 3][vr] = (__bf16)a.w;
            Vt[vc0 + 4][vr] = (__bf16)b.x; Vt[vc0 + 5][vr] = (__bf16)b.y;
            Vt[vc0 + 6][vr] = (__bf16)b.z; Vt[vc0 + 7][vr] = (__bf16)b.w;
        }
        if (w == 0) __builtin_amdgcn_s_wait_tensorcnt(0);
        __syncthreads();                       // K + V tiles visible

        // ---- K fragments from LDS fp32 ----
        v16bf kf[2][2];
#pragma unroll
        for (int t = 0; t < 2; ++t) {
#pragma unroll
            for (int f = 0; f < 2; ++f) {
                const float* p = &Ksf[t * 16 + laneM][f * 32 + hi * 8];
                float4 x0 = *(const float4*)(p);
                float4 x1 = *(const float4*)(p + 4);
                float4 x2 = *(const float4*)(p + 16);
                float4 x3 = *(const float4*)(p + 20);
                BF16x16 tt;
                put8(tt.v, 0, x0, x1);
                put8(tt.v, 8, x2, x3);
                kf[t][f] = tt.v;
            }
        }

        v8f s0 = {}, s1 = {};
        s0 = wmma_bf16(qf[0], kf[0][0], s0);
        s0 = wmma_bf16(qf[1], kf[0][1], s0);
        s1 = wmma_bf16(qf[0], kf[1][0], s1);
        s1 = wmma_bf16(qf[1], kf[1][1], s1);

        // ---- causal mask + streaming softmax ----
        float p0[8], p1[8], mc[8];
#pragma unroll
        for (int r = 0; r < 8; ++r) {
            int gq  = q0 + r + hi * 8;
            int gk0 = j0 + laneM;
            float a0 = (gk0 <= gq)      ? s0[r] : -3.0e38f;
            float a1 = (gk0 + 16 <= gq) ? s1[r] : -3.0e38f;
            s0[r] = a0; s1[r] = a1;
            mc[r] = fmaxf(a0, a1);
        }
#pragma unroll
        for (int off = 1; off < 16; off <<= 1)
#pragma unroll
            for (int r = 0; r < 8; ++r)
                mc[r] = fmaxf(mc[r], __shfl_xor(mc[r], off, 32));

        float alpha[8], rs[8];
#pragma unroll
        for (int r = 0; r < 8; ++r) {
            float mn = fmaxf(mrow[r], mc[r]);
            alpha[r] = __expf(mrow[r] - mn);
            mrow[r]  = mn;
            p0[r]    = __expf(s0[r] - mn);
            p1[r]    = __expf(s1[r] - mn);
            rs[r]    = p0[r] + p1[r];
        }
#pragma unroll
        for (int off = 1; off < 16; off <<= 1)
#pragma unroll
            for (int r = 0; r < 8; ++r)
                rs[r] += __shfl_xor(rs[r], off, 32);
#pragma unroll
        for (int r = 0; r < 8; ++r) lrow[r] = lrow[r] * alpha[r] + rs[r];
#pragma unroll
        for (int n = 0; n < 4; ++n)
#pragma unroll
            for (int r = 0; r < 8; ++r) o[n][r] *= alpha[r];

        // ---- re-layout P (f32 C-layout) -> bf16 A-fragment via LDS ----
        asm volatile("s_wait_dscnt 0x0" ::: "memory");   // prior Sp reads done
#pragma unroll
        for (int r = 0; r < 8; ++r) {
            int Mr = r + hi * 8;
            Sp[w][Mr][laneM]      = (__bf16)p0[r];
            Sp[w][Mr][16 + laneM] = (__bf16)p1[r];
        }
        asm volatile("s_wait_dscnt 0x0" ::: "memory");   // Sp stores visible
        __builtin_amdgcn_wave_barrier();

        v16bf pf;
        {
            const __bf16* p = &Sp[w][laneM][hi * 8];
            BF16x16 t;
            t.q[0] = *(const u32x4*)p;
            t.q[1] = *(const u32x4*)(p + 16);
            pf = t.v;
        }

        // ---- O += P @ V (vectorized B-fragments from transposed Vt) ----
#pragma unroll
        for (int n = 0; n < 4; ++n) {
            const __bf16* p = &Vt[n * 16 + laneM][hi * 8];
            BF16x16 t;
            t.q[0] = *(const u32x4*)p;
            t.q[1] = *(const u32x4*)(p + 16);
            o[n] = wmma_bf16(pf, t.v, o[n]);
        }
    }

    // ---- normalize and store ----
    float inv[8];
#pragma unroll
    for (int r = 0; r < 8; ++r) inv[r] = 1.0f / lrow[r];
    float* op = Om + (size_t)bh * Tt * HD;
#pragma unroll
    for (int n = 0; n < 4; ++n)
#pragma unroll
        for (int r = 0; r < 8; ++r)
            op[(size_t)(q0 + r + hi * 8) * HD + n * 16 + laneM] = o[n][r] * inv[r];
}

// ---------------------------------------------------------------------------
// Fused residual-add + LayerNorm. proj may be null (plain LN). out may alias res.
// ---------------------------------------------------------------------------
__global__ __launch_bounds__(256) void ln_kernel(
    const float* __restrict__ res, const float* __restrict__ proj,
    const float* __restrict__ g, const float* __restrict__ b,
    float* __restrict__ out)
{
    __shared__ float red[16];
    const int row = blockIdx.x;
    const int tid = threadIdx.x;
    const int c   = tid * 4;

    float4 v = *(const float4*)(res + (size_t)row * Dm + c);
    if (proj) {
        float4 p = *(const float4*)(proj + (size_t)row * Dm + c);
        v.x += p.x; v.y += p.y; v.z += p.z; v.w += p.w;
    }
    float s  = v.x + v.y + v.z + v.w;
    float sq = v.x * v.x + v.y * v.y + v.z * v.z + v.w * v.w;
#pragma unroll
    for (int off = 1; off < 32; off <<= 1) {
        s  += __shfl_xor(s, off, 32);
        sq += __shfl_xor(sq, off, 32);
    }
    if ((tid & 31) == 0) { red[tid >> 5] = s; red[8 + (tid >> 5)] = sq; }
    __syncthreads();
    float ts = 0.f, tq = 0.f;
#pragma unroll
    for (int i = 0; i < 8; ++i) { ts += red[i]; tq += red[8 + i]; }
    float mean = ts * (1.0f / Dm);
    float var  = tq * (1.0f / Dm) - mean * mean;
    float rstd = rsqrtf(var + 1e-5f);

    float4 gg = *(const float4*)(g + c);
    float4 bb = *(const float4*)(b + c);
    float4 r;
    r.x = (v.x - mean) * rstd * gg.x + bb.x;
    r.y = (v.y - mean) * rstd * gg.y + bb.y;
    r.z = (v.z - mean) * rstd * gg.z + bb.z;
    r.w = (v.w - mean) * rstd * gg.w + bb.w;
    *(float4*)(out + (size_t)row * Dm + c) = r;
}

// ---------------------------------------------------------------------------
// logits = X[:, 1::3, :] @ w_out  ([4096,1024] x [1024,18] -> tiny)
// ---------------------------------------------------------------------------
__global__ __launch_bounds__(256) void logits_kernel(
    const float* __restrict__ X, const float* __restrict__ Wout,
    float* __restrict__ out)
{
    int idx = blockIdx.x * blockDim.x + threadIdx.x;  // B*SEQ*ACT
    if (idx >= Bsz * SEQ * ACTN) return;
    int a = idx % ACTN;
    int s = (idx / ACTN) % SEQ;
    int b = idx / (ACTN * SEQ);
    const float* xr = X + (size_t)(b * Tt + 3 * s + 1) * Dm;
    float acc = 0.f;
#pragma unroll 4
    for (int d = 0; d < Dm; ++d) acc += xr[d] * Wout[d * ACTN + a];
    out[idx] = acc;
}

// ---------------------------------------------------------------------------
// Host orchestration
// ---------------------------------------------------------------------------
extern "C" void kernel_launch(void* const* d_in, const int* in_sizes, int n_in,
                              void* d_out, int out_size, void* d_ws, size_t ws_size,
                              hipStream_t stream)
{
    (void)in_sizes; (void)n_in; (void)out_size; (void)ws_size;

    const float* state   = (const float*)d_in[0];
    const int*   action  = (const int*)  d_in[1];
    const float* rtg     = (const float*)d_in[2];
    const int*   tstep   = (const int*)  d_in[3];
    const float* wS      = (const float*)d_in[4];
    const float* bS      = (const float*)d_in[5];
    const float* embA    = (const float*)d_in[6];
    const float* wR      = (const float*)d_in[7];
    const float* bR      = (const float*)d_in[8];
    const float* pos_emb = (const float*)d_in[9];
    const float* gpe     = (const float*)d_in[10];
    const float* w_out   = (const float*)d_in[11];
    const float* an_g    = (const float*)d_in[12];
    const float* an_b    = (const float*)d_in[13];

    char* wsp = (char*)d_ws;
    size_t off = 0;
    auto alloc = [&](size_t bytes) -> void* {
        void* p = wsp + off;
        off += (bytes + 255) & ~(size_t)255;
        return p;
    };
    float*  X   = (float*)alloc((size_t)Mtok * Dm * 4);
    float*  Qb  = (float*)alloc((size_t)Mtok * Dm * 4);
    float*  Kb  = (float*)alloc((size_t)Mtok * Dm * 4);
    float*  Vb  = (float*)alloc((size_t)Mtok * Dm * 4);
    float*  Ob  = (float*)alloc((size_t)Mtok * Dm * 4);
    float*  Pb  = (float*)alloc((size_t)Mtok * Dm * 4);
    float*  Hb  = (float*)alloc((size_t)Mtok * FFN * 4);
    __bf16* Wbf = (__bf16*)alloc((size_t)FFN * Dm * 2);

    // 1) token + positional embedding
    {
        int n = Bsz * SEQ * Dm;
        embed_kernel<<<(n + 255) / 256, 256, 0, stream>>>(
            state, action, rtg, tstep, wS, bS, embA, wR, bR, pos_emb, gpe, X);
    }

    auto cvt = [&](const float* src, int K, int N) {
        cvt_bf16_t_kernel<<<2048, 256, 0, stream>>>(src, Wbf, K, N);
    };
    auto gemm = [&](const float* A, const float* bias, float* C,
                    int M, int N, int K, int relu) {
        dim3 grid(N / 128, M / 128);
        if (relu)
            gemm_wmma_kernel<1><<<grid, 256, 0, stream>>>(A, Wbf, bias, C, M, N, K);
        else
            gemm_wmma_kernel<0><<<grid, 256, 0, stream>>>(A, Wbf, bias, C, M, N, K);
    };

    for (int blk = 0; blk < NBLK; ++blk) {
        const int base = 14 + blk * 16;
        const float* wq  = (const float*)d_in[base + 0];
        const float* bq  = (const float*)d_in[base + 1];
        const float* wk  = (const float*)d_in[base + 2];
        const float* bk  = (const float*)d_in[base + 3];
        const float* wv  = (const float*)d_in[base + 4];
        const float* bv  = (const float*)d_in[base + 5];
        const float* wo  = (const float*)d_in[base + 6];
        const float* bo  = (const float*)d_in[base + 7];
        const float* w1  = (const float*)d_in[base + 8];
        const float* b1  = (const float*)d_in[base + 9];
        const float* w2  = (const float*)d_in[base + 10];
        const float* b2  = (const float*)d_in[base + 11];
        const float* n1g = (const float*)d_in[base + 12];
        const float* n1b = (const float*)d_in[base + 13];
        const float* n2g = (const float*)d_in[base + 14];
        const float* n2b = (const float*)d_in[base + 15];

        cvt(wq, Dm, Dm);  gemm(X, bq, Qb, Mtok, Dm, Dm, 0);
        cvt(wk, Dm, Dm);  gemm(X, bk, Kb, Mtok, Dm, Dm, 0);
        cvt(wv, Dm, Dm);  gemm(X, bv, Vb, Mtok, Dm, Dm, 0);

        attn_kernel<<<dim3(BH, Tt / 128), 256, 0, stream>>>(Qb, Kb, Vb, Ob);

        cvt(wo, Dm, Dm);  gemm(Ob, bo, Pb, Mtok, Dm, Dm, 0);
        ln_kernel<<<Mtok, 256, 0, stream>>>(X, Pb, n1g, n1b, X);

        cvt(w1, Dm, FFN); gemm(X, b1, Hb, Mtok, FFN, Dm, 1);
        cvt(w2, FFN, Dm); gemm(Hb, b2, Pb, Mtok, Dm, FFN, 0);
        ln_kernel<<<Mtok, 256, 0, stream>>>(X, Pb, n2g, n2b, X);
    }

    // final LayerNorm + output projection
    ln_kernel<<<Mtok, 256, 0, stream>>>(X, nullptr, an_g, an_b, X);
    {
        int n = Bsz * SEQ * ACTN;
        logits_kernel<<<(n + 255) / 256, 256, 0, stream>>>(X, w_out, (float*)d_out);
    }
}
